// Attention1d_46042049413167
// MI455X (gfx1250) — matrix operand order repaired and checked by reading, not compiled, not run on hardware
//
#include <hip/hip_runtime.h>
#include <hip/hip_bf16.h>

using half_t = _Float16;
typedef __attribute__((ext_vector_type(16))) _Float16 v16h;
typedef __attribute__((ext_vector_type(8)))  float    v8f;
typedef unsigned int u32x4 __attribute__((ext_vector_type(4)));
typedef int          i32x4 __attribute__((ext_vector_type(4)));
typedef int          i32x8 __attribute__((ext_vector_type(8)));

#define GLOBAL_AS __attribute__((address_space(1)))
#define LDS_AS    __attribute__((address_space(3)))

// ---- feature probes (keep fallbacks so every toolchain compiles) ----
#if defined(__has_builtin)
#  if __has_builtin(__builtin_amdgcn_tensor_load_to_lds) && __has_builtin(__builtin_amdgcn_s_wait_tensorcnt)
#    define HAVE_TDM 1
#  endif
#  if __has_builtin(__builtin_amdgcn_global_load_async_to_lds_b128)
#    define HAVE_ASYNC 1
#  endif
#endif
#ifndef HAVE_TDM
#define HAVE_TDM 0
#endif
#ifndef HAVE_ASYNC
#define HAVE_ASYNC 0
#endif

// ---- problem constants ----
constexpr int BB    = 16;    // batch
constexpr int DIMc  = 512;   // model dim
constexpr int Lc    = 1024;  // sequence length (RES)
constexpr int NHc   = 8;     // heads
constexpr int KDc   = 32;    // key dim
constexpr int Dc    = 128;   // value dim per head
constexpr int DHc   = 1024;  // D * heads
constexpr int NHKD  = 256;   // key_dim * heads
constexpr int HQKV  = 1536;  // DH + 2*NH_KD
constexpr float SCALEF = 0.176776695296636881f; // 32^-0.5

// =====================================================================
// CDNA5 data-movement helpers
// =====================================================================
static __device__ __forceinline__ void async_b128(const half_t* g, half_t* l) {
#if HAVE_ASYNC
    // prototype (from hipcc diagnostic): (v4i addrspace(1)*, v4i addrspace(3)*, imm, imm)
    __builtin_amdgcn_global_load_async_to_lds_b128(
        (GLOBAL_AS i32x4*)g, (LDS_AS i32x4*)l, 0, 0);
#else
    (void)g; (void)l;
#endif
}

static __device__ __forceinline__ void wait_async0() {
#if HAVE_ASYNC
#  if __has_builtin(__builtin_amdgcn_s_wait_asynccnt)
    __builtin_amdgcn_s_wait_asynccnt(0);
#  else
    asm volatile("s_wait_asynccnt 0x0" ::: "memory");
#  endif
#endif
}

// 2-D TDM tile load: tileY rows of tileX elements (2B each), row stride
// stride0 elements in memory, packed into LDS with pitch 40 halves
// (pad_interval=3 -> every 16 DWORDs, pad_amount=3 -> insert 4 DWORDs).
static __device__ __forceinline__ void tdm_load_tile_2d(
    const half_t* gsrc, unsigned ldsOffBytes, unsigned tileX, unsigned tileY,
    unsigned long long dim0, unsigned long long dim1, unsigned long long stride0) {
#if HAVE_TDM
    unsigned long long ga = (unsigned long long)(size_t)gsrc;
    u32x4 g0 = {0u, 0u, 0u, 0u};
    g0[0] = 1u;                                               // count=1 (valid)
    g0[1] = ldsOffBytes;                                      // lds_addr
    g0[2] = (unsigned)ga;                                     // global_addr[31:0]
    g0[3] = ((unsigned)(ga >> 32) & 0x01FFFFFFu) | (2u << 30);// [56:32] | type=2
    i32x8 g1 = {0, 0, 0, 0, 0, 0, 0, 0};
    g1[0] = (int)((1u << 16)        // data_size = 2 bytes
                | (1u << 20)        // pad_enable
                | (3u << 22)        // pad_interval: 16 DWORDs
                | (3u << 25));      // pad_amount: 4 DWORDs
    g1[1] = (int)((dim0 & 0xFFFFu) << 16);                            // tensor_dim0[15:0]
    g1[2] = (int)(((dim0 >> 16) & 0xFFFFu) | ((dim1 & 0xFFFFu) << 16));// dim0[31:16]|dim1[15:0]
    g1[3] = (int)(((dim1 >> 16) & 0xFFFFu) | (tileX << 16));          // dim1[31:16]|tile_dim0
    g1[4] = (int)(tileY & 0xFFFFu);                                   // tile_dim1 (tile_dim2=0)
    g1[5] = (int)(stride0 & 0xFFFFFFFFull);                           // dim0_stride[31:0]
    g1[6] = (int)((stride0 >> 32) & 0xFFFFull);                       // dim0_stride[47:32]
    g1[7] = 0;
    i32x4 zz = {0, 0, 0, 0};
#  if __clang_major__ >= 23
    i32x8 z8 = {0, 0, 0, 0, 0, 0, 0, 0};
    __builtin_amdgcn_tensor_load_to_lds(g0, g1, zz, zz, z8, 0);
#  else
    __builtin_amdgcn_tensor_load_to_lds(g0, g1, zz, zz, 0);
#  endif
#else
    (void)gsrc; (void)ldsOffBytes; (void)tileX; (void)tileY;
    (void)dim0; (void)dim1; (void)stride0;
#endif
}

static __device__ __forceinline__ void wait_tensor0() {
#if HAVE_TDM
    __builtin_amdgcn_s_wait_tensorcnt(0);
#endif
}

// =====================================================================
// WMMA fragment loaders (this code's contiguous K order, shared by A and B)
//   A (MxK = 16x32): lane&15 -> M row; g=lane/16 selects K-halves:
//     elem e -> K = e + 16g
//   B (KxN = 32x16): stored transposed in LDS as [N][K]:
//     lane&15 -> N col; elem e -> K = e + 16g
// =====================================================================
__device__ __forceinline__ v16h load_A_frag(const half_t* lds, int rowBase, int pitch) {
    int lane = threadIdx.x & 31;
    int g = lane >> 4, m = lane & 15;
    const half_t* p = lds + (rowBase + m) * pitch;
    v16h a;
#pragma unroll
    for (int e = 0; e < 16; ++e) a[e] = p[e + 16 * g];
    return a;
}

__device__ __forceinline__ v16h load_B_frag(const half_t* lds, int colBase, int pitch) {
    int lane = threadIdx.x & 31;
    int g = lane >> 4, n = lane & 15;
    const half_t* p = lds + (colBase + n) * pitch + 16 * g;
    v16h b;
#pragma unroll
    for (int e = 0; e < 16; ++e) b[e] = p[e];
    return b;
}

__device__ __forceinline__ v8f wmma_f16(v16h a, v16h b, v8f c) {
    return __builtin_amdgcn_wmma_f32_16x16x32_f16(false, a, false, b, (short)0, c, false, false);
}

// =====================================================================
// prep: fold BN params:  y*alpha + beta  ==  (y-m)*g/sqrt(v+eps) + b
// =====================================================================
__global__ void prep_bn_kernel(const float* __restrict__ g, const float* __restrict__ b,
                               const float* __restrict__ m, const float* __restrict__ v,
                               float* __restrict__ alpha, float* __restrict__ beta, int n) {
    int i = blockIdx.x * blockDim.x + threadIdx.x;
    if (i < n) {
        float inv = g[i] * rsqrtf(v[i] + 1e-5f);
        alpha[i] = inv;
        beta[i]  = b[i] - m[i] * inv;
    }
}

__global__ void f32_to_f16_kernel(const float* __restrict__ src, half_t* __restrict__ dst, int n) {
    for (int i = blockIdx.x * blockDim.x + threadIdx.x; i < n; i += gridDim.x * blockDim.x)
        dst[i] = (half_t)src[i];
}

// =====================================================================
// Tiled WMMA GEMM + BN epilogue.
//   out[o,l] = alpha[o] * (sum_c W[o,c] * X[c,l]) + beta[o]
// mode 0 (qkv): split o into q(f32) | k(f16) | v(f16);  mode 1 (proj): f32
// 128x128 tile, K-step 32, 8 waves: wave grid 4(o) x 2(l), each 32x64.
// A tile is staged by the Tensor Data Mover (padded 2-D tile) when available.
// =====================================================================
__global__ __launch_bounds__(256)
void gemm_bn_kernel(const half_t* __restrict__ W, const half_t* __restrict__ Xall,
                    const float* __restrict__ alpha, const float* __restrict__ beta,
                    int Cin, int L, int mode,
                    float* __restrict__ outF,
                    float* __restrict__ outQ, half_t* __restrict__ outKH, half_t* __restrict__ outVH) {
    __shared__ half_t sA[128 * 40];   // [o][k], pitch 40
    __shared__ half_t sBt[128 * 40];  // [l][k]  (B transposed)

    const int tid  = threadIdx.x;
    const int lane = tid & 31;
    const int wave = tid >> 5;
    const int wo = wave & 3, wl = wave >> 2;
    const int b = blockIdx.z;
    const int oBase = blockIdx.y * 128;
    const int lBase = blockIdx.x * 128;
    const half_t* X = Xall + (size_t)b * Cin * L;

    v8f acc[2][4];
#pragma unroll
    for (int i = 0; i < 2; ++i)
#pragma unroll
        for (int j = 0; j < 4; ++j) acc[i][j] = (v8f){};

    const int ar = tid >> 1, ac0 = (tid & 1) * 16;        // A loader coords
    const int xk = tid >> 3, xl0 = (tid & 7) * 16;        // X loader coords

    for (int k0 = 0; k0 < Cin; k0 += 32) {
        // ---- A tile: 128 x 32 (row-major, pitch-40 LDS) ----
#if HAVE_TDM
        if (wave == 0) {
            tdm_load_tile_2d(W + (size_t)oBase * Cin + k0,
                             (unsigned)(size_t)(LDS_AS half_t*)&sA[0],
                             32u, 128u,
                             (unsigned long long)Cin, (unsigned long long)HQKV,
                             (unsigned long long)Cin);
        }
#elif HAVE_ASYNC
        {
            const half_t* src = W + (size_t)(oBase + ar) * Cin + k0 + ac0;
            async_b128(src,     &sA[ar * 40 + ac0]);
            async_b128(src + 8, &sA[ar * 40 + ac0 + 8]);
        }
#else
        {
            const half_t* src = W + (size_t)(oBase + ar) * Cin + k0 + ac0;
            uint4 t0 = *(const uint4*)(src);
            uint4 t1 = *(const uint4*)(src + 8);
            *(uint4*)&sA[ar * 40 + ac0]     = t0;
            *(uint4*)&sA[ar * 40 + ac0 + 8] = t1;
        }
#endif
        // ---- X tile: 32 x 128 -> LDS transposed [l][k] ----
        {
            const half_t* src = X + (size_t)(k0 + xk) * L + lBase + xl0;
            half_t tmp[16];
            *(uint4*)&tmp[0] = *(const uint4*)(src);
            *(uint4*)&tmp[8] = *(const uint4*)(src + 8);
#pragma unroll
            for (int i = 0; i < 16; ++i) sBt[(xl0 + i) * 40 + xk] = tmp[i];
            if (k0 + 32 < Cin) __builtin_prefetch(src + 32, 0, 0);
        }
#if HAVE_TDM
        if (wave == 0) wait_tensor0();
#elif HAVE_ASYNC
        wait_async0();
#endif
        __syncthreads();

        v16h af[2], bf[4];
#pragma unroll
        for (int i = 0; i < 2; ++i) af[i] = load_A_frag(sA, wo * 32 + i * 16, 40);
#pragma unroll
        for (int j = 0; j < 4; ++j) bf[j] = load_B_frag(sBt, wl * 64 + j * 16, 40);
#pragma unroll
        for (int i = 0; i < 2; ++i)
#pragma unroll
            for (int j = 0; j < 4; ++j)
                acc[i][j] = wmma_f16(af[i], bf[j], acc[i][j]);
        __syncthreads();
    }

    // BN epilogue + routed stores
    const int g = lane >> 4, nl = lane & 15;
#pragma unroll
    for (int i = 0; i < 2; ++i) {
#pragma unroll
        for (int j = 0; j < 4; ++j) {
            const int l = lBase + wl * 64 + j * 16 + nl;
#pragma unroll
            for (int r = 0; r < 8; ++r) {
                const int o = oBase + wo * 32 + i * 16 + r + 8 * g;
                float val = alpha[o] * acc[i][j][r] + beta[o];
                if (mode == 1) {
                    outF[((size_t)b * 512 + o) * L + l] = val;
                } else {
                    if (o < NHKD)          outQ [((size_t)b * NHKD + o) * L + l] = val;
                    else if (o < 2 * NHKD) outKH[((size_t)b * NHKD + (o - NHKD)) * L + l] = (half_t)val;
                    else                   outVH[((size_t)b * DHc + (o - 2 * NHKD)) * L + l] = (half_t)val;
                }
            }
        }
    }
}

// =====================================================================
// depthwise conv3 (pad 1) + BN, f32 in -> f16 out
// =====================================================================
__global__ void dwconv_bn_kernel(const float* __restrict__ q, const float* __restrict__ w_dw,
                                 const float* __restrict__ alpha, const float* __restrict__ beta,
                                 half_t* __restrict__ qh) {
    int idx = blockIdx.x * blockDim.x + threadIdx.x;
    if (idx >= BB * NHKD * Lc) return;
    int l = idx & (Lc - 1);
    int c = (idx >> 10) & (NHKD - 1);
    const float* row = q + (size_t)(idx >> 10) * Lc;
    float xm1 = (l > 0)      ? row[l - 1] : 0.f;
    float x0  = row[l];
    float xp1 = (l < Lc - 1) ? row[l + 1] : 0.f;
    float v = w_dw[c * 3 + 0] * xm1 + w_dw[c * 3 + 1] * x0 + w_dw[c * 3 + 2] * xp1;
    qh[idx] = (half_t)(alpha[c] * v + beta[c]);
}

// =====================================================================
// Fused flash attention per (b,h): S = Q^T K * scale + bias(|n-m|),
// online softmax, O = V P^T, ReLU, f16 out. 128 query rows / block,
// 8 waves x 16 rows, 32-key chunks (K=32 == one WMMA).
// V tile is staged with async global->LDS copies when available.
// =====================================================================
__global__ __launch_bounds__(256)
void attn_kernel(const half_t* __restrict__ qh, const half_t* __restrict__ kh,
                 const half_t* __restrict__ vh, const float* __restrict__ bias,
                 half_t* __restrict__ oh) {
    __shared__ half_t sQ[128 * 40];      // [n][d]
    __shared__ half_t sK[32 * 40];       // [m][d]
    __shared__ half_t sV[128 * 40];      // [d][m]
    __shared__ half_t sP[8][16 * 40];    // per-wave P^T staging [n][m]
    __shared__ float  sRow[8][16];       // per-wave row broadcast

    const int tid  = threadIdx.x;
    const int lane = tid & 31;
    const int wave = tid >> 5;
    const int g = lane >> 4, nl = lane & 15;
    const int b = blockIdx.z, h = blockIdx.y, blk = blockIdx.x;
    const int nb = wave * 16;

    // stage Q^T (128 x 32) once
    {
        int n = tid >> 1, d0 = (tid & 1) * 16;
        const half_t* src = qh + ((size_t)b * NHKD + h * KDc + d0) * Lc + blk * 128 + n;
#pragma unroll
        for (int i = 0; i < 16; ++i) sQ[n * 40 + d0 + i] = src[(size_t)i * Lc];
    }
    __syncthreads();

    v8f oacc[8];
#pragma unroll
    for (int di = 0; di < 8; ++di) oacc[di] = (v8f){};
    float run_max[8], run_sum[8];
#pragma unroll
    for (int r = 0; r < 8; ++r) { run_max[r] = -1e30f; run_sum[r] = 0.f; }

    const float* biasH = bias + h * Lc;

    for (int m0 = 0; m0 < Lc; m0 += 32) {
        // V tile 128x32 -> [d][m]
        {
            int d = tid >> 1, mi0 = (tid & 1) * 16;
            const half_t* src = vh + ((size_t)b * DHc + h * Dc + d) * Lc + m0 + mi0;
#if HAVE_ASYNC
            async_b128(src,     &sV[d * 40 + mi0]);
            async_b128(src + 8, &sV[d * 40 + mi0 + 8]);
#else
            uint4 t0 = *(const uint4*)src;
            uint4 t1 = *(const uint4*)(src + 8);
            *(uint4*)&sV[d * 40 + mi0]     = t0;
            *(uint4*)&sV[d * 40 + mi0 + 8] = t1;
#endif
        }
        // K tile 32x32 -> [m][d] (transpose: manual)
        {
            int d = tid >> 3, mi0 = (tid & 7) * 4;
            const half_t* src = kh + ((size_t)b * NHKD + h * KDc + d) * Lc + m0 + mi0;
            half_t tmp[4];
            *(uint2*)&tmp[0] = *(const uint2*)src;
#pragma unroll
            for (int i = 0; i < 4; ++i) sK[(mi0 + i) * 40 + d] = tmp[i];
        }
        wait_async0();
        __syncthreads();

        // S = Q^T K  (two 16-col tiles)
        v16h aq = load_A_frag(sQ, nb, 40);
        v8f s[2];
#pragma unroll
        for (int j = 0; j < 2; ++j) {
            v16h bk = load_B_frag(sK, j * 16, 40);
            s[j] = wmma_f16(aq, bk, (v8f){});
        }

        // scale + positional bias + tile row-max
        float tmax[8];
#pragma unroll
        for (int r = 0; r < 8; ++r) tmax[r] = -1e30f;
#pragma unroll
        for (int j = 0; j < 2; ++j) {
#pragma unroll
            for (int r = 0; r < 8; ++r) {
                int n_g = blk * 128 + nb + r + 8 * g;
                int m_g = m0 + j * 16 + nl;
                int d = n_g - m_g; if (d < 0) d = -d;
                float val = s[j][r] * SCALEF + biasH[d];
                s[j][r] = val;
                tmax[r] = fmaxf(tmax[r], val);
            }
        }
#pragma unroll
        for (int off = 1; off < 16; off <<= 1)
#pragma unroll
            for (int r = 0; r < 8; ++r)
                tmax[r] = fmaxf(tmax[r], __shfl_xor(tmax[r], off, 32));

        // online softmax update
        float corr[8], psum[8];
#pragma unroll
        for (int r = 0; r < 8; ++r) {
            float nm = fmaxf(run_max[r], tmax[r]);
            corr[r] = __expf(run_max[r] - nm);
            run_max[r] = nm;
            psum[r] = 0.f;
        }
#pragma unroll
        for (int j = 0; j < 2; ++j)
#pragma unroll
            for (int r = 0; r < 8; ++r) {
                float p = __expf(s[j][r] - run_max[r]);
                s[j][r] = p;
                psum[r] += p;
            }
#pragma unroll
        for (int off = 1; off < 16; off <<= 1)
#pragma unroll
            for (int r = 0; r < 8; ++r)
                psum[r] += __shfl_xor(psum[r], off, 32);
#pragma unroll
        for (int r = 0; r < 8; ++r) run_sum[r] = run_sum[r] * corr[r] + psum[r];

        // broadcast correction from S-layout rows to O-layout columns
        if (nl == 0) {
#pragma unroll
            for (int r = 0; r < 8; ++r) sRow[wave][r + 8 * g] = corr[r];
        }
        float cn = sRow[wave][nl];
#pragma unroll
        for (int di = 0; di < 8; ++di)
#pragma unroll
            for (int r = 0; r < 8; ++r) oacc[di][r] *= cn;

        // P^T -> per-wave LDS [n][m]
        half_t* sPw = &sP[wave][0];
#pragma unroll
        for (int j = 0; j < 2; ++j)
#pragma unroll
            for (int r = 0; r < 8; ++r)
                sPw[(r + 8 * g) * 40 + j * 16 + nl] = (half_t)s[j][r];

        // O += V * P^T   (8 d-tiles, K=32 in one WMMA)
        v16h bp = load_B_frag(sPw, 0, 40);
#pragma unroll
        for (int di = 0; di < 8; ++di) {
            v16h av = load_A_frag(sV, di * 16, 40);
            oacc[di] = wmma_f16(av, bp, oacc[di]);
        }
        __syncthreads();
    }

    // normalize, ReLU, store f16
    if (nl == 0) {
#pragma unroll
        for (int r = 0; r < 8; ++r) sRow[wave][r + 8 * g] = 1.0f / run_sum[r];
    }
    float invn = sRow[wave][nl];
    const int n_g = blk * 128 + nb + nl;
#pragma unroll
    for (int di = 0; di < 8; ++di) {
#pragma unroll
        for (int r = 0; r < 8; ++r) {
            int c = h * Dc + di * 16 + r + 8 * g;
            float val = fmaxf(oacc[di][r] * invn, 0.0f);
            oh[((size_t)b * DHc + c) * Lc + n_g] = (half_t)val;
        }
    }
}

// =====================================================================
// launcher
// =====================================================================
extern "C" void kernel_launch(void* const* d_in, const int* in_sizes, int n_in,
                              void* d_out, int out_size, void* d_ws, size_t ws_size,
                              hipStream_t stream) {
    const float* x       = (const float*)d_in[0];
    const float* w_qkv   = (const float*)d_in[1];
    const float* qkv_g   = (const float*)d_in[2];
    const float* qkv_b   = (const float*)d_in[3];
    const float* qkv_m   = (const float*)d_in[4];
    const float* qkv_v   = (const float*)d_in[5];
    const float* w_dw    = (const float*)d_in[6];
    const float* dw_g    = (const float*)d_in[7];
    const float* dw_b    = (const float*)d_in[8];
    const float* dw_m    = (const float*)d_in[9];
    const float* dw_v    = (const float*)d_in[10];
    const float* w_proj  = (const float*)d_in[11];
    const float* proj_g  = (const float*)d_in[12];
    const float* proj_b  = (const float*)d_in[13];
    const float* proj_m  = (const float*)d_in[14];
    const float* proj_v  = (const float*)d_in[15];
    const float* ab      = (const float*)d_in[16];
    float* out = (float*)d_out;

    // workspace carve-up
    char* ws = (char*)d_ws;
    size_t off = 0;
    auto carve = [&](size_t bytes) { char* p = ws + off; off = (off + bytes + 255) & ~(size_t)255; return p; };
    half_t* xh     = (half_t*)carve((size_t)BB * DIMc * Lc * 2);
    half_t* wqkvh  = (half_t*)carve((size_t)HQKV * DIMc * 2);
    half_t* wprojh = (half_t*)carve((size_t)DIMc * DHc * 2);
    float*  a_qkv  = (float*)carve(HQKV * 4);
    float*  b_qkv  = (float*)carve(HQKV * 4);
    float*  a_dw   = (float*)carve(NHKD * 4);
    float*  b_dw   = (float*)carve(NHKD * 4);
    float*  a_pr   = (float*)carve(DIMc * 4);
    float*  b_pr   = (float*)carve(DIMc * 4);
    float*  qf     = (float*)carve((size_t)BB * NHKD * Lc * 4);
    half_t* khb    = (half_t*)carve((size_t)BB * NHKD * Lc * 2);
    half_t* vhb    = (half_t*)carve((size_t)BB * DHc * Lc * 2);
    half_t* qhb    = (half_t*)carve((size_t)BB * NHKD * Lc * 2);
    half_t* ohb    = (half_t*)carve((size_t)BB * DHc * Lc * 2);
    (void)ws_size; (void)in_sizes; (void)n_in; (void)out_size;

    // BN folding
    prep_bn_kernel<<<(HQKV + 255) / 256, 256, 0, stream>>>(qkv_g, qkv_b, qkv_m, qkv_v, a_qkv, b_qkv, HQKV);
    prep_bn_kernel<<<(NHKD + 255) / 256, 256, 0, stream>>>(dw_g, dw_b, dw_m, dw_v, a_dw, b_dw, NHKD);
    prep_bn_kernel<<<(DIMc + 255) / 256, 256, 0, stream>>>(proj_g, proj_b, proj_m, proj_v, a_pr, b_pr, DIMc);

    // f32 -> f16 packs
    f32_to_f16_kernel<<<4096, 256, 0, stream>>>(x, xh, BB * DIMc * Lc);
    f32_to_f16_kernel<<<1024, 256, 0, stream>>>(w_qkv, wqkvh, HQKV * DIMc);
    f32_to_f16_kernel<<<1024, 256, 0, stream>>>(w_proj, wprojh, DIMc * DHc);

    // qkv GEMM + BN, split q(f32)/k(f16)/v(f16)
    gemm_bn_kernel<<<dim3(Lc / 128, HQKV / 128, BB), 256, 0, stream>>>(
        wqkvh, xh, a_qkv, b_qkv, DIMc, Lc, /*mode=*/0, nullptr, qf, khb, vhb);

    // depthwise conv + BN on q
    dwconv_bn_kernel<<<(BB * NHKD * Lc + 255) / 256, 256, 0, stream>>>(qf, w_dw, a_dw, b_dw, qhb);

    // fused flash attention + ReLU
    attn_kernel<<<dim3(Lc / 128, NHc, BB), 256, 0, stream>>>(qhb, khb, vhb, ab, ohb);

    // proj GEMM + BN -> f32 output
    gemm_bn_kernel<<<dim3(Lc / 128, DIMc / 128, BB), 256, 0, stream>>>(
        wprojh, ohb, a_pr, b_pr, DHc, Lc, /*mode=*/1, out, nullptr, nullptr, nullptr);
}
